// CentroidPool_62852551409936
// MI455X (gfx1250) — matrix operand order repaired and checked by reading, not compiled, not run on hardware
//
#include <hip/hip_runtime.h>

// Problem constants (from the reference): latent [65536,128] f32, coords [4096,128] f32.
#define N_PTS 65536
#define K_CEN 4096
#define DIM   128
#define ROWS_PER_WG 128            // 8 waves x 16-row WMMA tiles
#define NTILES (K_CEN / 16)        // 256 centroid tiles

typedef __attribute__((ext_vector_type(16))) __bf16 v16bf;
typedef __attribute__((ext_vector_type(8)))  __bf16 v8bf;
typedef __attribute__((ext_vector_type(8)))  float  v8f;

// ---------------------------------------------------------------------------
// Kernel 1: split coords into bf16 hi/lo (bf16x3 decomposition) and compute
// per-centroid squared norms c2[k] in fp32. One 128-thread block per centroid.
// ---------------------------------------------------------------------------
__global__ __launch_bounds__(128) void prep_centroids(
    const float* __restrict__ coords,
    __bf16* __restrict__ chi, __bf16* __restrict__ clo,
    float* __restrict__ c2) {
  const int k = blockIdx.x;
  const int t = threadIdx.x;            // 0..127 == D index
  const float x = coords[k * DIM + t];
  const __bf16 h = (__bf16)x;           // RNE round to bf16
  chi[k * DIM + t] = h;
  clo[k * DIM + t] = (__bf16)(x - (float)h);

  // fp32 norm reduction (exact inputs; accuracy matters for argmin bias)
  float s = x * x;
  #pragma unroll
  for (int m = 16; m >= 1; m >>= 1) s += __shfl_xor(s, m, 32);
  __shared__ float red[4];
  if ((t & 31) == 0) red[t >> 5] = s;
  __syncthreads();
  if (t == 0) c2[k] = red[0] + red[1] + red[2] + red[3];
}

// ---------------------------------------------------------------------------
// Kernel 2: fused (c2 - 2*X·C^T) + running argmin over centroids.
// 256 threads = 8 waves; each wave owns a 16-row latent tile (A in VGPRs).
// The 16-centroid B tile (bf16 hi+lo, 8 KB) is staged once per WG into a
// double-buffered LDS tile shared by all 8 waves -> coords bf16 data is read
// from L2 exactly once per WG. Prefetches (B tile + c2) are issued AFTER the
// barrier so their loadcnt waits land a full compute phase later, overlapped
// with the 12 v_wmma_f32_16x16x32_bf16 (bf16x3 split: hi*hi + hi*lo + lo*hi).
// ---------------------------------------------------------------------------
__global__ __launch_bounds__(256) void argmin_wmma(
    const float* __restrict__ latent,
    const __bf16* __restrict__ chi, const __bf16* __restrict__ clo,
    const float* __restrict__ c2, float* __restrict__ out) {
  __shared__ __align__(32) __bf16 sAhi[ROWS_PER_WG * DIM];    // 32 KB
  __shared__ __align__(32) __bf16 sAlo[ROWS_PER_WG * DIM];    // 32 KB
  // [buf][fmt][centroid 0..15][D 0..127]
  __shared__ __align__(32) __bf16 sB[2][2][16][DIM];          // 16 KB

  const int tid  = threadIdx.x;
  const int row0 = blockIdx.x * ROWS_PER_WG;

  // Stage + split this block's 128 latent rows (read from HBM exactly once).
  for (int i = tid; i < ROWS_PER_WG * DIM; i += 256) {
    const float x = latent[row0 * DIM + i];
    const __bf16 h = (__bf16)x;
    sAhi[i] = h;
    sAlo[i] = (__bf16)(x - (float)h);
  }

  const int wave  = tid >> 5;
  const int lane  = tid & 31;
  const int lmod  = lane & 15;   // N column within tile / A row within tile
  const int lhalf = lane >> 4;   // lane-half selects K sub-ranges

  // Per-thread slice of the B tile prefetch: 16 contiguous bf16 (32 B).
  const int pf_fmt  = tid >> 7;          // 0=hi, 1=lo
  const int pf_cent = (tid & 127) >> 3;
  const int pf_dcol = (tid & 7) * 16;
  const __bf16* pf_base =
      (pf_fmt ? clo : chi) + pf_cent * DIM + pf_dcol;

  // Prefetch tile 0 (B slice) and c2 column 0 into registers.
  v16bf pf = *(const v16bf*)(pf_base);
  float cc = c2[lmod];

  __syncthreads();   // A staging complete

  // A fragments (16x32 bf16 layout): lane<16 holds K {0..7,16..23},
  // lane>=16 holds K {8..15,24..31} of its row. 4 chunks cover D=128.
  v16bf ahi[4], alo[4];
  {
    const int row = wave * 16 + lmod;
    #pragma unroll
    for (int c = 0; c < 4; ++c) {
      const int d0 = 32 * c + lhalf * 8;
      const v8bf h0 = *(const v8bf*)(sAhi + row * DIM + d0);
      const v8bf h1 = *(const v8bf*)(sAhi + row * DIM + d0 + 16);
      const v8bf l0 = *(const v8bf*)(sAlo + row * DIM + d0);
      const v8bf l1 = *(const v8bf*)(sAlo + row * DIM + d0 + 16);
      #pragma unroll
      for (int e = 0; e < 8; ++e) {
        ahi[c][e] = h0[e]; ahi[c][e + 8] = h1[e];
        alo[c][e] = l0[e]; alo[c][e + 8] = l1[e];
      }
    }
  }

  // Running argmin. C/D layout: acc[j] is row M=j+8*lhalf, col N=kc+lmod.
  float best[8];
  int   bestk[8];
  #pragma unroll
  for (int j = 0; j < 8; ++j) { best[j] = 3.4e38f; bestk[j] = 0; }

  for (int it = 0; it < NTILES; ++it) {
    const int buf = it & 1;
    // Publish the prefetched tile to LDS for all 8 waves. The loadcnt wait
    // for 'pf' lands here -- one full compute phase after it was issued.
    *(v16bf*)(&sB[buf][pf_fmt][pf_cent][pf_dcol]) = pf;

    __syncthreads();   // only needs dscnt==0; single barrier/iter is safe
                       // with double buffering.

    // Issue next-tile prefetches AFTER the barrier so they overlap the
    // WMMA/ds_load compute phase below (clamped uniform index, no branch).
    const int nt = (it + 1 < NTILES) ? (it + 1) : it;
    pf = *(const v16bf*)(pf_base + (size_t)nt * 16 * DIM);
    const float cc_next = c2[nt * 16 + lmod];

    const int n = it * 16 + lmod;   // this lane's centroid column
    // B fragments (32x16 bf16 layout): lane<16 holds K {0..7,16..23}, lane>=16
    // holds K {8..15,24..31} of column n (matches A fragment order).
    v16bf bhi[4], blo[4];
    #pragma unroll
    for (int c = 0; c < 4; ++c) {
      const v8bf bh0 = *(const v8bf*)(&sB[buf][0][lmod][32 * c + lhalf * 8]);
      const v8bf bh1 = *(const v8bf*)(&sB[buf][0][lmod][32 * c + lhalf * 8 + 16]);
      const v8bf bl0 = *(const v8bf*)(&sB[buf][1][lmod][32 * c + lhalf * 8]);
      const v8bf bl1 = *(const v8bf*)(&sB[buf][1][lmod][32 * c + lhalf * 8 + 16]);
      #pragma unroll
      for (int e = 0; e < 8; ++e) {
        bhi[c][e] = bh0[e]; bhi[c][e + 8] = bh1[e];
        blo[c][e] = bl0[e]; blo[c][e + 8] = bl1[e];
      }
    }

    v8f acc = {};
    #pragma unroll
    for (int c = 0; c < 4; ++c) {
      acc = __builtin_amdgcn_wmma_f32_16x16x32_bf16(
          false, ahi[c], false, bhi[c], (short)0, acc, false, false);
      acc = __builtin_amdgcn_wmma_f32_16x16x32_bf16(
          false, ahi[c], false, blo[c], (short)0, acc, false, false);
      acc = __builtin_amdgcn_wmma_f32_16x16x32_bf16(
          false, alo[c], false, bhi[c], (short)0, acc, false, false);
    }

    #pragma unroll
    for (int j = 0; j < 8; ++j) {
      const float s = fmaf(-2.0f, acc[j], cc);   // x^2 constant per row: dropped
      if (s < best[j]) { best[j] = s; bestk[j] = n; }
    }
    cc = cc_next;
  }

  // Min+index reduction across the 16 lanes holding each row's columns.
  #pragma unroll
  for (int j = 0; j < 8; ++j) {
    float b  = best[j];
    int   bk = bestk[j];
    #pragma unroll
    for (int m = 1; m < 16; m <<= 1) {
      const float ob  = __shfl_xor(b, m, 32);
      const int   obk = __shfl_xor(bk, m, 32);
      if (ob < b || (ob == b && obk < bk)) { b = ob; bk = obk; }
    }
    if (lmod == 0) {
      out[row0 + wave * 16 + lhalf * 8 + j] = (float)bk;
    }
  }
}

// ---------------------------------------------------------------------------
extern "C" void kernel_launch(void* const* d_in, const int* in_sizes, int n_in,
                              void* d_out, int out_size, void* d_ws, size_t ws_size,
                              hipStream_t stream) {
  const float* latent = (const float*)d_in[0];
  const float* coords = (const float*)d_in[1];
  float* out = (float*)d_out;   // argmin indices written as float values

  // Workspace layout: chi[4096*128] bf16 | clo[4096*128] bf16 | c2[4096] f32
  // = 1 MB + 1 MB + 16 KB ≈ 2.02 MB.
  char* ws = (char*)d_ws;
  __bf16* chi = (__bf16*)(ws);
  __bf16* clo = (__bf16*)(ws + (size_t)K_CEN * DIM * sizeof(__bf16));
  float*  c2  = (float*)(ws + (size_t)2 * K_CEN * DIM * sizeof(__bf16));

  prep_centroids<<<K_CEN, 128, 0, stream>>>(coords, chi, clo, c2);
  argmin_wmma<<<N_PTS / ROWS_PER_WG, 256, 0, stream>>>(latent, chi, clo, c2, out);

  (void)in_sizes; (void)n_in; (void)out_size; (void)ws_size;
}
